// GraphGenerativeModel_20710332301755
// MI455X (gfx1250) — matrix operand order repaired and checked by reading, not compiled, not run on hardware
//
#include <hip/hip_runtime.h>

#define N_TOTAL    65536
#define NUM_GRAPHS 1024
#define PER_GRAPH  64
#define NUM_E      1048576
#define C          128        // IN_C == HID_C == OUT_C
#define FC_N       65536
#define LDW        65         // padded LDS row stride in dwords (bank-conflict-free)

typedef __attribute__((ext_vector_type(16))) __bf16 v16bf;
typedef __attribute__((ext_vector_type(8)))  float  v8f;

union Frag { unsigned u[8]; v16bf v; };

// round-to-nearest-even f32 -> bf16, packed two per dword
__device__ __forceinline__ unsigned pack_bf16x2(float a, float b) {
  unsigned ua = __builtin_bit_cast(unsigned, a);
  unsigned ub = __builtin_bit_cast(unsigned, b);
  ua += 0x7FFFu + ((ua >> 16) & 1u);
  ub += 0x7FFFu + ((ub >> 16) & 1u);
  return (ua >> 16) | (ub & 0xFFFF0000u);
}

// A-matrix 16x32 bf16 frag per ISA 7.12.2: lane&15 = M row; lanes>=16 shift K by 8.
// p0 points at LDS word (row_tile*16)*LDW + kstep*16.
__device__ __forceinline__ v16bf load_a_frag(const unsigned* p0, int lane) {
  const int m  = lane & 15;
  const int kb = (lane >> 4) << 2;          // 0 or 4 dwords (K 0 / K 8)
  const unsigned* p = p0 + m * LDW;
  Frag f;
#pragma unroll
  for (int j = 0; j < 4; ++j) { f.u[j] = p[kb + j]; f.u[4 + j] = p[8 + kb + j]; }
  return f.v;
}

// B-matrix 32x16 bf16 frag: lane&15 = N col; lanes<16 -> K0-7,16-23, lanes>=16 -> K8-15,24-31.
// LDS holds B column-major: ldsB[n*LDW + k/2].
__device__ __forceinline__ v16bf load_b_frag(const unsigned* p0, int lane) {
  const int n  = lane & 15;
  const int kb = (lane >> 4) << 2;          // 0 or 4 dwords
  const unsigned* p = p0 + n * LDW + kb;
  Frag f;
#pragma unroll
  for (int j = 0; j < 4; ++j) { f.u[j] = p[j]; f.u[4 + j] = p[8 + j]; }
  return f.v;
}

// out[row0..row0+63][n0..n0+127] = A[rows][0..127] @ B[0..127][cols] (+bias)
// A row stride = C (128). B row stride = ldb. out row stride = ldout.
__global__ void __launch_bounds__(256)
gemm_bf16_wmma(const float* __restrict__ A, const float* __restrict__ B,
               const float* __restrict__ bias, float* __restrict__ out,
               int ldb, int ldout) {
  __shared__ unsigned ldsA[64  * LDW];      // 64 rows x 128 K (bf16 pairs)
  __shared__ unsigned ldsB[128 * LDW];      // 128 cols x 128 K, column-major
  const int t    = threadIdx.x;
  const int row0 = blockIdx.y * 64;
  const int n0   = blockIdx.x * 128;

  for (int i = t; i < 64 * 64; i += 256) {  // stage A, coalesced float2 reads
    int r = i >> 6, w = i & 63;
    const float* p = A + (long)(row0 + r) * C + 2 * w;
    ldsA[r * LDW + w] = pack_bf16x2(p[0], p[1]);
  }
  for (int i = t; i < 64 * 128; i += 256) { // stage B, consecutive lanes -> consecutive n
    int w = i >> 7, n = i & 127;
    float f0 = B[(long)(2 * w)     * ldb + n0 + n];
    float f1 = B[(long)(2 * w + 1) * ldb + n0 + n];
    ldsB[n * LDW + w] = pack_bf16x2(f0, f1);
  }
  __syncthreads();

  const int wave = t >> 5, lane = t & 31;
  const unsigned* bBase = ldsB + (wave * 16) * LDW;   // one N tile per wave

  // epilogue constants (hoisted: independent of the M-tile loop)
  const int n    = lane & 15;
  const int mb   = (lane >> 4) * 8;
  const int ncol = n0 + wave * 16 + n;
  const float bv = bias ? bias[ncol] : 0.0f;

#pragma unroll
  for (int mt = 0; mt < 4; ++mt) {
    v8f c = {};
#pragma unroll
    for (int ks = 0; ks < 4; ++ks) {                  // K = 128 = 4 x 32
      v16bf a = load_a_frag(ldsA + (mt * 16) * LDW + ks * 16, lane);
      v16bf b = load_b_frag(bBase + ks * 16, lane);
      c = __builtin_amdgcn_wmma_f32_16x16x32_bf16(false, a, false, b,
                                                  (short)0, c, false, false);
    }
    // C/D layout: VGPR r -> M = r (lanes 0-15) or 8+r (lanes 16-31), N = lane&15
#pragma unroll
    for (int r = 0; r < 8; ++r)
      out[(long)(row0 + mt * 16 + mb + r) * ldout + ncol] = c[r] + bv;
  }
}

__global__ void init_deg(float* __restrict__ deg) {
  deg[blockIdx.x * 256 + threadIdx.x] = 1.0f;         // self loop
}

__global__ void deg_accum(const long long* __restrict__ ei, float* __restrict__ deg) {
  int e = blockIdx.x * 256 + threadIdx.x;
  unsafeAtomicAdd(&deg[(int)ei[NUM_E + e]], 1.0f);
}

__global__ void deg_to_dinv(float* __restrict__ deg) {
  int i = blockIdx.x * 256 + threadIdx.x;
  deg[i] = rsqrtf(deg[i]);                            // deg >= 1 always
}

__global__ void zero_f(float* __restrict__ p) {
  p[blockIdx.x * 256 + threadIdx.x] = 0.0f;
}

// 16 threads per edge, 8 channels each: agg[dst] += hw[src] * dinv[src]*dinv[dst]
__global__ void __launch_bounds__(256)
scatter_edges(const float* __restrict__ hw, const long long* __restrict__ ei,
              const float* __restrict__ dinv, float* __restrict__ agg) {
  unsigned idx  = blockIdx.x * 256 + threadIdx.x;
  unsigned e    = idx >> 4;
  unsigned part = idx & 15;
  int s = (int)ei[e];
  int d = (int)ei[NUM_E + e];
  float w = dinv[s] * dinv[d];
  const float* src = hw  + (long)s * C + part * 8;
  float*       dst = agg + (long)d * C + part * 8;
#pragma unroll
  for (int c = 0; c < 8; ++c) unsafeAtomicAdd(&dst[c], src[c] * w);
}

// h = relu(agg + hw*dinv^2 (self loop) + bias), in place in agg
__global__ void self_bias_relu(float* __restrict__ agg, const float* __restrict__ hw,
                               const float* __restrict__ dinv,
                               const float* __restrict__ bias) {
  int i = blockIdx.x * 256 + threadIdx.x;
  int node = i >> 7, c = i & 127;
  float di = dinv[node];
  float v = agg[i] + hw[i] * di * di + bias[c];
  agg[i] = v > 0.0f ? v : 0.0f;
}

// One block per graph. Stage the 64x128 f32 node block into LDS with the
// gfx1250 async global->LDS DMA (ASYNCcnt-tracked, no VGPR round trip),
// then reduce the 64 rows per channel.
__global__ void __launch_bounds__(256)
pool_mean_async(const float* __restrict__ h, float* __restrict__ pooled) {
  __shared__ __align__(16) float tile[PER_GRAPH * C];   // 32 KB raw f32 copy
  const int g = blockIdx.x;
  const int t = threadIdx.x;
  const float* src = h + (size_t)g * PER_GRAPH * C;

  // 8192 floats / (256 threads * 4 floats per b128) = 8 async issues per thread.
  // Low 32 bits of a generic LDS pointer are the wave-relative LDS byte offset.
#pragma unroll
  for (int it = 0; it < 8; ++it) {
    int elt = (it * 256 + t) * 4;
    unsigned     ldsaddr = (unsigned)(size_t)(&tile[elt]);
    const float* gaddr   = src + elt;
    asm volatile("global_load_async_to_lds_b128 %0, %1, off"
                 :: "v"(ldsaddr), "v"(gaddr) : "memory");
  }
  asm volatile("s_wait_asynccnt 0x0" ::: "memory");
  __syncthreads();

  if (t < C) {
    float s = 0.0f;
#pragma unroll 4
    for (int k = 0; k < PER_GRAPH; ++k) s += tile[k * C + t];
    pooled[(size_t)g * C + t] = s * (1.0f / PER_GRAPH);
  }
}

extern "C" void kernel_launch(void* const* d_in, const int* in_sizes, int n_in,
                              void* d_out, int out_size, void* d_ws, size_t ws_size,
                              hipStream_t stream) {
  const float*     x   = (const float*)d_in[0];
  const long long* ei  = (const long long*)d_in[1];   // int64 [2, E]
  // d_in[2] = batch (node i -> i/64, recomputed implicitly)
  const float*     W1  = (const float*)d_in[3];
  const float*     b1  = (const float*)d_in[4];
  const float*     W2  = (const float*)d_in[5];
  const float*     b2  = (const float*)d_in[6];
  const float*     Wfc = (const float*)d_in[7];
  const float*     bfc = (const float*)d_in[8];
  float* out = (float*)d_out;

  float* deg    = (float*)d_ws;                       // 65536 (becomes dinv)
  float* hw     = deg + N_TOTAL;                      // 65536*128
  float* agg    = hw  + (size_t)N_TOTAL * C;          // 65536*128
  float* pooled = agg + (size_t)N_TOTAL * C;          // 1024*128

  // degrees + symmetric norm
  init_deg   <<<N_TOTAL / 256, 256, 0, stream>>>(deg);
  deg_accum  <<<NUM_E   / 256, 256, 0, stream>>>(ei, deg);
  deg_to_dinv<<<N_TOTAL / 256, 256, 0, stream>>>(deg);

  // layer 1: hw = x @ W1 ; agg = scatter ; h1 = relu(agg + self + b1) (in agg)
  gemm_bf16_wmma<<<dim3(1, N_TOTAL / 64), 256, 0, stream>>>(x, W1, nullptr, hw, C, C);
  zero_f        <<<(N_TOTAL * C) / 256, 256, 0, stream>>>(agg);
  scatter_edges <<<(NUM_E * 16) / 256, 256, 0, stream>>>(hw, ei, deg, agg);
  self_bias_relu<<<(N_TOTAL * C) / 256, 256, 0, stream>>>(agg, hw, deg, b1);

  // layer 2 (h1 lives in agg; reuse hw for hw2, then recycle agg)
  gemm_bf16_wmma<<<dim3(1, N_TOTAL / 64), 256, 0, stream>>>(agg, W2, nullptr, hw, C, C);
  zero_f        <<<(N_TOTAL * C) / 256, 256, 0, stream>>>(agg);
  scatter_edges <<<(NUM_E * 16) / 256, 256, 0, stream>>>(hw, ei, deg, agg);
  self_bias_relu<<<(N_TOTAL * C) / 256, 256, 0, stream>>>(agg, hw, deg, b2);

  // pool (async DMA to LDS) + FC (output written directly with bias)
  pool_mean_async<<<NUM_GRAPHS, 256, 0, stream>>>(agg, pooled);
  gemm_bf16_wmma<<<dim3(FC_N / 128, NUM_GRAPHS / 64), 256, 0, stream>>>(
      pooled, Wfc, bfc, out, FC_N, FC_N);
}
